// VGG_SMALL_1W1A_17222818857396
// MI455X (gfx1250) — 8-bit matrix operand order repaired and checked by reading, compiled, not run on hardware
//
#include <hip/hip_runtime.h>
#include <hip/hip_bf16.h>
#include <cstdint>

#define BN_EPS 1e-5f

typedef __attribute__((ext_vector_type(8))) int v8i;
typedef __attribute__((ext_vector_type(4))) int v4i;

static __device__ __forceinline__ float bn_apply(float v, float s1, float s2,
                                                 float invN, float gg, float bb) {
    float m   = s1 * invN;
    float var = fmaxf(s2 * invN - m * m, 0.f);
    return (v - m) * rsqrtf(var + BN_EPS) * gg + bb;
}
static __device__ __forceinline__ float clip1(float v) {
    return fminf(1.f, fmaxf(-1.f, v));
}

__global__ __launch_bounds__(128)
void k_binconv_wmma(const int8_t* __restrict__ act,
                    const int8_t* __restrict__ wgt,
                    int* __restrict__ out,
                    int H, int W, int Cin, int Cout)
{
    const int tid   = threadIdx.x;
    const int lane  = tid & 31;
    const int wave  = tid >> 5;
    const int tilesN = Cout >> 6;
    const int tileM  = blockIdx.x / tilesN;
    const int tileN  = blockIdx.x - tileM * tilesN;
    const int m0     = tileM * 64 + wave * 16;
    const int row    = m0 + (lane & 15);
    const int HW  = H * W;
    const int n   = row / HW;
    const int hw  = row - n * HW;
    const int h   = hw / W;
    const int w   = hw - h * W;
    const int Hp = H + 2, Wp = W + 2;
    const size_t actBase = (((size_t)n * Hp + h) * Wp + w) * Cin;
    const int half = lane >> 4;
    const int col  = lane & 15;
    const int wrow = 9 * Cin;
    const int kSteps = Cin >> 6;
    const int S = 9 * kSteps;

    __shared__ v4i Bs[2][256];
    const size_t wTileBase = (size_t)tileN * 64 * wrow;

    v8i acc[4] = {};

    int rS = 0, kcS = 0;
    int rC = 0, kcC = 0;

    {
        const size_t kb = (size_t)rS * Cin + (size_t)kcS * 64;
        #pragma unroll
        for (int u = 0; u < 2; ++u) {
            const int q  = tid + u * 128;
            const int co = q >> 2, so = (q & 3) * 16;
            Bs[0][q] = *(const v4i*)(wgt + wTileBase + (size_t)co * wrow + kb + so);
        }
        ++kcS; if (kcS == kSteps) { kcS = 0; ++rS; }
    }
    __syncthreads();

    for (int s = 0; s < S; ++s) {
        if (s + 1 < S) {
            const int bufn = (s + 1) & 1;
            const size_t kb = (size_t)rS * Cin + (size_t)kcS * 64;
            #pragma unroll
            for (int u = 0; u < 2; ++u) {
                const int q  = tid + u * 128;
                const int co = q >> 2, so = (q & 3) * 16;
                Bs[bufn][q] = *(const v4i*)(wgt + wTileBase + (size_t)co * wrow + kb + so);
            }
            ++kcS; if (kcS == kSteps) { kcS = 0; ++rS; }
        }

        const int ky = rC / 3, kx = rC - ky * 3;
        const int8_t* ap = act + actBase + ((size_t)ky * Wp + kx) * Cin
                         + (size_t)kcC * 64 + half * 16;
        __builtin_prefetch(ap + 64, 0, 1);

        uint64_t a0 = *(const uint64_t*)(ap);
        uint64_t a1 = *(const uint64_t*)(ap + 8);
        uint64_t a2 = *(const uint64_t*)(ap + 32);
        uint64_t a3 = *(const uint64_t*)(ap + 40);
        v8i A = {};
        A[0] = (int)(uint32_t)a0; A[1] = (int)(a0 >> 32);
        A[2] = (int)(uint32_t)a1; A[3] = (int)(a1 >> 32);
        A[4] = (int)(uint32_t)a2; A[5] = (int)(a2 >> 32);
        A[6] = (int)(uint32_t)a3; A[7] = (int)(a3 >> 32);

        const int buf = s & 1;
        #pragma unroll
        for (int b = 0; b < 4; ++b) {
            const int cq = (b * 16 + col) * 4 + half;
            v4i blo = Bs[buf][cq];
            v4i bhi = Bs[buf][cq + 2];
            v8i B = {};
            B[0] = blo[0]; B[1] = blo[1]; B[2] = blo[2]; B[3] = blo[3];
            B[4] = bhi[0]; B[5] = bhi[1]; B[6] = bhi[2]; B[7] = bhi[3];
            acc[b] = __builtin_amdgcn_wmma_i32_16x16x64_iu8(
                true, A, true, B, acc[b], false, false);
        }

        ++kcC; if (kcC == kSteps) { kcC = 0; ++rC; }
        __syncthreads();
    }

    #pragma unroll
    for (int b = 0; b < 4; ++b) {
        const int co = tileN * 64 + b * 16 + col;
        #pragma unroll
        for (int j = 0; j < 8; ++j) {
            const int m = m0 + half * 8 + j;
            out[(size_t)m * Cout + co] = acc[b][j];
        }
    }
}

__global__ void k_conv0(const float* __restrict__ x, const float* __restrict__ w0,
                        float* __restrict__ out, long total)
{
    long i = (long)blockIdx.x * blockDim.x + threadIdx.x;
    if (i >= total) return;
    int  q = (int)(i & 1023);
    long r = i >> 10;
    int  k = (int)(r & 127);
    int  n = (int)(r >> 7);
    int  h = q >> 5, w = q & 31;
    float acc = 0.f;
    #pragma unroll
    for (int c = 0; c < 3; ++c)
        #pragma unroll
        for (int ky = 0; ky < 3; ++ky) {
            int hy = h + ky - 1;
            if (hy < 0 || hy > 31) continue;
            #pragma unroll
            for (int kx = 0; kx < 3; ++kx) {
                int wx = w + kx - 1;
                if (wx < 0 || wx > 31) continue;
                acc += x[(((size_t)n * 3 + c) * 32 + hy) * 32 + wx]
                     * w0[((k * 3 + c) * 3 + ky) * 3 + kx];
            }
        }
    out[i] = acc;
}

__global__ void k_zero(float* __restrict__ p, long n)
{
    long i = (long)blockIdx.x * blockDim.x + threadIdx.x;
    if (i < n) p[i] = 0.f;
}

__global__ void k_stats_f32(const float* __restrict__ src, float* __restrict__ stats,
                            int Nn, int C, int HW)
{
    const int c = blockIdx.x;
    const int tid = threadIdx.x;
    const long total = (long)Nn * HW;
    float s1 = 0.f, s2 = 0.f;
    for (long p = (long)blockIdx.y * blockDim.x + tid; p < total;
         p += (long)gridDim.y * blockDim.x) {
        long n = p / HW, q = p - n * HW;
        float v = src[((size_t)n * C + c) * HW + q];
        s1 += v; s2 += v * v;
    }
    __shared__ float r1[256], r2[256];
    r1[tid] = s1; r2[tid] = s2; __syncthreads();
    for (int o = 128; o > 0; o >>= 1) {
        if (tid < o) { r1[tid] += r1[tid + o]; r2[tid] += r2[tid + o]; }
        __syncthreads();
    }
    if (tid == 0) { atomicAdd(&stats[2 * c], r1[0]); atomicAdd(&stats[2 * c + 1], r2[0]); }
}

__global__ void k_stats_i32(const int* __restrict__ src, const float* __restrict__ alpha,
                            float* __restrict__ stats, int Np, int C)
{
    const int c = blockIdx.x;
    const int tid = threadIdx.x;
    const float a = alpha[c];
    float s1 = 0.f, s2 = 0.f;
    for (long p = (long)blockIdx.y * blockDim.x + tid; p < Np;
         p += (long)gridDim.y * blockDim.x) {
        float v = a * (float)src[(size_t)p * C + c];
        s1 += v; s2 += v * v;
    }
    __shared__ float r1[256], r2[256];
    r1[tid] = s1; r2[tid] = s2; __syncthreads();
    for (int o = 128; o > 0; o >>= 1) {
        if (tid < o) { r1[tid] += r1[tid + o]; r2[tid] += r2[tid + o]; }
        __syncthreads();
    }
    if (tid == 0) { atomicAdd(&stats[2 * c], r1[0]); atomicAdd(&stats[2 * c + 1], r2[0]); }
}

__global__ void k_bn0(const float* __restrict__ conv, const float* __restrict__ stats,
                      const float* __restrict__ g, const float* __restrict__ b,
                      float* __restrict__ x0, long total, float invN)
{
    long i = (long)blockIdx.x * blockDim.x + threadIdx.x;
    if (i >= total) return;
    int  q = (int)(i & 1023);
    long r = i >> 10;
    int  c = (int)(r & 255);
    long n = r >> 8;
    int cc = c & 127;
    float v = conv[((size_t)n * 128 + cc) * 1024 + q];
    x0[i] = clip1(bn_apply(v, stats[2 * cc], stats[2 * cc + 1], invN, g[cc], b[cc]));
}

__global__ void k_signpack(const float* __restrict__ x, int8_t* __restrict__ out,
                           long total, int Cprev, int Cin, int H, int W)
{
    long i = (long)blockIdx.x * blockDim.x + threadIdx.x;
    if (i >= total) return;
    int  c  = (int)(i % Cin);
    long r  = i / Cin;
    int  wp = (int)(r % (W + 2)); r /= (W + 2);
    int  hp = (int)(r % (H + 2));
    long n  = r / (H + 2);
    int8_t s = 0;
    if (hp >= 1 && hp <= H && wp >= 1 && wp <= W) {
        float v = x[(((size_t)n * Cprev + c) * H + (hp - 1)) * W + (wp - 1)];
        s = (v > 0.f) ? (int8_t)1 : ((v < 0.f) ? (int8_t)-1 : (int8_t)0);
    }
    out[i] = s;
}

__global__ void k_wprep(const float* __restrict__ w, int8_t* __restrict__ ws,
                        float* __restrict__ alpha, int Cin)
{
    const int co = blockIdx.x;
    const int t  = threadIdx.x;
    const int total = Cin * 9;
    float s = 0.f;
    for (int j = t; j < total; j += 256) {
        int c = j / 9, r = j - c * 9;
        float v = w[(size_t)co * total + j];
        s += fabsf(v);
        ws[(size_t)co * total + (size_t)r * Cin + c] =
            (v > 0.f) ? (int8_t)1 : ((v < 0.f) ? (int8_t)-1 : (int8_t)0);
    }
    __shared__ float red[256];
    red[t] = s; __syncthreads();
    for (int o = 128; o > 0; o >>= 1) { if (t < o) red[t] += red[t + o]; __syncthreads(); }
    if (t == 0) alpha[co] = red[0] / (float)total;
}

__global__ void k_maxpool_i32(const int* __restrict__ in, int* __restrict__ out,
                              long total, int H, int W, int C)
{
    long i = (long)blockIdx.x * blockDim.x + threadIdx.x;
    if (i >= total) return;
    const int W2 = W >> 1, H2 = H >> 1;
    int  c  = (int)(i % C);
    long r  = i / C;
    int  w2 = (int)(r % W2); r /= W2;
    int  h2 = (int)(r % H2);
    long n  = r / H2;
    size_t base = (((size_t)n * H + 2 * h2) * W + 2 * w2) * C + c;
    int m = in[base];
    m = max(m, in[base + C]);
    m = max(m, in[base + (size_t)W * C]);
    m = max(m, in[base + (size_t)W * C + C]);
    out[i] = m;
}

__global__ void k_v3_epilogue(const int* __restrict__ conv, const float* __restrict__ alpha,
                              const float* __restrict__ stats,
                              const float* __restrict__ g, const float* __restrict__ b,
                              const float* __restrict__ xprev, float* __restrict__ xnext,
                              long total, int Cout, int Cprev, int inpl, int Ct,
                              int H, int W, int H2, int W2, int pooling, float invNp)
{
    long i = (long)blockIdx.x * blockDim.x + threadIdx.x;
    if (i >= total) return;
    int  w2 = (int)(i % W2);
    long r  = i / W2;
    int  h2 = (int)(r % H2); r /= H2;
    int  c  = (int)(r % Ct);
    long n  = r / Ct;
    const int sc = (c & 1) * (Ct >> 1) + (c >> 1);
    float val;
    if (sc < Cout) {
        float v = alpha[sc] *
                  (float)conv[(((size_t)n * H2 + h2) * W2 + w2) * Cout + sc];
        val = clip1(bn_apply(v, stats[2 * sc], stats[2 * sc + 1], invNp, g[sc], b[sc]));
    } else {
        int si = sc - Cout;
        if (pooling) {
            const float* p = &xprev[(((size_t)n * Cprev + inpl + si) * H + 2 * h2) * W + 2 * w2];
            val = 0.25f * (p[0] + p[1] + p[W] + p[W + 1]);
        } else {
            val = xprev[(((size_t)n * Cprev + si) * H + h2) * W + w2];
        }
    }
    xnext[i] = val;
}

__global__ void k_dense_epi(const int* __restrict__ conv, const float* __restrict__ alpha,
                            const float* __restrict__ stats,
                            const float* __restrict__ g, const float* __restrict__ b,
                            const float* __restrict__ xprev, float* __restrict__ feat,
                            long total, float invNp)
{
    long i = (long)blockIdx.x * blockDim.x + threadIdx.x;
    if (i >= total) return;
    int  f  = (int)(i & 8191);
    long n  = i >> 13;
    int  c  = f >> 4;
    int  hw = f & 15;
    int  h  = hw >> 2, w = hw & 3;
    float v = alpha[c] * (float)conv[(((size_t)n * 4 + h) * 4 + w) * 512 + c];
    float val = bn_apply(v, stats[2 * c], stats[2 * c + 1], invNp, g[c], b[c]);
    val += xprev[((size_t)n * 1024 + c) * 16 + hw];
    feat[i] = clip1(val);
}

__global__ void k_bn1d_stats(const float* __restrict__ feat, float* __restrict__ mv, int N)
{
    int f = blockIdx.x * blockDim.x + threadIdx.x;
    if (f >= 8192) return;
    float s1 = 0.f, s2 = 0.f;
    for (int n = 0; n < N; ++n) {
        float v = feat[(size_t)n * 8192 + f];
        s1 += v; s2 += v * v;
    }
    float m   = s1 / (float)N;
    float var = fmaxf(s2 / (float)N - m * m, 0.f);
    mv[2 * f]     = m;
    mv[2 * f + 1] = rsqrtf(var + BN_EPS);
}

__global__ void k_fc(const float* __restrict__ feat, const float* __restrict__ mv,
                     const float* __restrict__ gbn, const float* __restrict__ bbn,
                     const float* __restrict__ wfc, const float* __restrict__ bfc,
                     float* __restrict__ out, int N)
{
    int gt   = blockIdx.x * blockDim.x + threadIdx.x;
    int warp = gt >> 5;
    int lane = threadIdx.x & 31;
    if (warp >= N * 10) return;
    int n = warp / 10, j = warp - n * 10;
    float s = 0.f;
    for (int f = lane; f < 8192; f += 32) {
        float xv = (feat[(size_t)n * 8192 + f] - mv[2 * f]) * mv[2 * f + 1] * gbn[f] + bbn[f];
        s += xv * wfc[(size_t)j * 8192 + f];
    }
    for (int o = 16; o > 0; o >>= 1) s += __shfl_down(s, o, 32);
    if (lane == 0) out[(size_t)n * 10 + j] = s + bfc[j];
}

extern "C" void kernel_launch(void* const* d_in, const int* in_sizes, int n_in,
                              void* d_out, int out_size, void* d_ws, size_t ws_size,
                              hipStream_t stream)
{
    (void)n_in; (void)out_size; (void)ws_size;
    const float* x   = (const float*)d_in[0];
    const float* W0  = (const float*)d_in[1];
    const float* G0  = (const float*)d_in[2];
    const float* B0  = (const float*)d_in[3];
    const float* Wv[5] = {(const float*)d_in[4],  (const float*)d_in[7],
                          (const float*)d_in[10], (const float*)d_in[13],
                          (const float*)d_in[16]};
    const float* Gv[5] = {(const float*)d_in[5],  (const float*)d_in[8],
                          (const float*)d_in[11], (const float*)d_in[14],
                          (const float*)d_in[17]};
    const float* Bv[5] = {(const float*)d_in[6],  (const float*)d_in[9],
                          (const float*)d_in[12], (const float*)d_in[15],
                          (const float*)d_in[18]};
    const float* WD  = (const float*)d_in[19];
    const float* GD  = (const float*)d_in[20];
    const float* BD  = (const float*)d_in[21];
    const float* GBN = (const float*)d_in[22];
    const float* BBN = (const float*)d_in[23];
    const float* WFC = (const float*)d_in[24];
    const float* BFC = (const float*)d_in[25];
    float* out = (float*)d_out;

    const int N = in_sizes[0] / (3 * 32 * 32);
    const int TB = 256;

    char* base = (char*)d_ws;
    size_t off = 0;
    auto carve = [&](size_t bytes) -> char* {
        char* p = base + off;
        off += (bytes + 255) & ~(size_t)255;
        return p;
    };
    float*  X0    = (float*) carve((size_t)N * 256 * 1024 * sizeof(float));
    float*  X1    = (float*) carve((size_t)N * 256 * 1024 * sizeof(float));
    int*    CONV  = (int*)   carve((size_t)N * 128 * 1024 * sizeof(int));
    int*    POOL  = (int*)   carve((size_t)N * 128 * 256  * sizeof(int));
    int8_t* SIGN  = (int8_t*)carve((size_t)N * 34 * 34 * 128);
    int8_t* WGT   = (int8_t*)carve((size_t)512 * 9 * 1024);
    float*  ALPHA = (float*) carve(512 * sizeof(float));
    float*  STATS = (float*) carve(2 * 512 * sizeof(float));
    float*  MV    = (float*) carve(2 * 8192 * sizeof(float));
    float*  FEAT  = (float*) carve((size_t)N * 8192 * sizeof(float));

    {
        long t0 = (long)N * 128 * 1024;
        float* CONVF = (float*)CONV;
        k_conv0<<<dim3((unsigned)((t0 + TB - 1) / TB)), TB, 0, stream>>>(x, W0, CONVF, t0);
        k_zero<<<dim3(1), TB, 0, stream>>>(STATS, 2 * 128);
        k_stats_f32<<<dim3(128, 64), TB, 0, stream>>>(CONVF, STATS, N, 128, 1024);
        long t1 = (long)N * 256 * 1024;
        k_bn0<<<dim3((unsigned)((t1 + TB - 1) / TB)), TB, 0, stream>>>(
            CONVF, STATS, G0, B0, X0, t1, 1.f / ((float)N * 1024.f));
    }

    float* xc = X0; float* xn = X1;
    int Cprev = 256, H = 32, Wd = 32;
    const int inplv[5] = {128, 128, 256, 256, 512};
    const int coutv[5] = {128, 256, 256, 512, 512};
    const int poolv[5] = {1, 0, 1, 0, 1};
    for (int bi = 0; bi < 5; ++bi) {
        const int inpl = inplv[bi], Cout = coutv[bi], pool = poolv[bi];
        long ts = (long)N * (H + 2) * (Wd + 2) * inpl;
        k_signpack<<<dim3((unsigned)((ts + TB - 1) / TB)), TB, 0, stream>>>(
            xc, SIGN, ts, Cprev, inpl, H, Wd);
        k_wprep<<<dim3(Cout), TB, 0, stream>>>(Wv[bi], WGT, ALPHA, inpl);
        const int Npix = N * H * Wd;
        k_binconv_wmma<<<dim3((unsigned)((Npix / 64) * (Cout / 64))), 128, 0, stream>>>(
            SIGN, WGT, CONV, H, Wd, inpl, Cout);
        const int H2 = pool ? H / 2 : H, W2 = pool ? Wd / 2 : Wd;
        const int* cb = CONV;
        if (pool) {
            long tp = (long)N * H2 * W2 * Cout;
            k_maxpool_i32<<<dim3((unsigned)((tp + TB - 1) / TB)), TB, 0, stream>>>(
                CONV, POOL, tp, H, Wd, Cout);
            cb = POOL;
        }
        k_zero<<<dim3(4), TB, 0, stream>>>(STATS, 2 * Cout);
        k_stats_i32<<<dim3(Cout, 32), TB, 0, stream>>>(cb, ALPHA, STATS, N * H2 * W2, Cout);
        const int Ct = Cout + (pool ? (Cprev - inpl) : Cprev);
        long te = (long)N * Ct * H2 * W2;
        k_v3_epilogue<<<dim3((unsigned)((te + TB - 1) / TB)), TB, 0, stream>>>(
            cb, ALPHA, STATS, Gv[bi], Bv[bi], xc, xn, te,
            Cout, Cprev, inpl, Ct, H, Wd, H2, W2, pool,
            1.f / ((float)N * H2 * W2));
        float* t = xc; xc = xn; xn = t;
        Cprev = Ct; H = H2; Wd = W2;
    }

    {
        long ts = (long)N * 6 * 6 * 1024;
        k_signpack<<<dim3((unsigned)((ts + TB - 1) / TB)), TB, 0, stream>>>(
            xc, SIGN, ts, 1024, 1024, 4, 4);
        k_wprep<<<dim3(512), TB, 0, stream>>>(WD, WGT, ALPHA, 1024);
        k_binconv_wmma<<<dim3((unsigned)(((N * 16) / 64) * (512 / 64))), 128, 0, stream>>>(
            SIGN, WGT, CONV, 4, 4, 1024, 512);
        k_zero<<<dim3(4), TB, 0, stream>>>(STATS, 2 * 512);
        k_stats_i32<<<dim3(512, 16), TB, 0, stream>>>(CONV, ALPHA, STATS, N * 16, 512);
        long tf = (long)N * 8192;
        k_dense_epi<<<dim3((unsigned)((tf + TB - 1) / TB)), TB, 0, stream>>>(
            CONV, ALPHA, STATS, GD, BD, xc, FEAT, tf, 1.f / ((float)N * 16.f));
        k_bn1d_stats<<<dim3(8192 / TB), TB, 0, stream>>>(FEAT, MV, N);
        const int warps = N * 10;
        k_fc<<<dim3((unsigned)((warps * 32 + TB - 1) / TB)), TB, 0, stream>>>(
            FEAT, MV, GBN, BBN, WFC, BFC, out, N);
    }
}
